// DiscreteNASLayer_68453188764134
// MI455X (gfx1250) — compile-verified, matrix operand order checked by reading, not run on hardware
//
#include <hip/hip_runtime.h>

#define DD  128
#define HH  8
#define HDm 16
#define DD2 256

typedef __attribute__((ext_vector_type(16))) __bf16          v16bf;
typedef __attribute__((ext_vector_type(8)))  float           v8f;
typedef unsigned short ushort_t;

// ---------------- helpers ----------------
__device__ __forceinline__ ushort_t f2bf(float f) {
  unsigned u = __float_as_uint(f);
  u += 0x7fffu + ((u >> 16) & 1u);            // round-to-nearest-even
  return (ushort_t)(u >> 16);
}
__device__ __forceinline__ int f2ord(float f) {       // monotone float->int
  int b = __float_as_int(f);
  return b >= 0 ? b : (b ^ 0x7fffffff);
}
__device__ __forceinline__ float ord2f(int o) {
  return __int_as_float(o >= 0 ? o : (o ^ 0x7fffffff));
}

union frag16 { uint4 q[2]; v16bf v; };

// ---------------- WMMA GEMM, bf16 in / f32 out ----------------
// A: bf16 row-major [M x K].  Bpk: bf16 pre-packed per-(ktile,ntile,lane) fragments.
// One wave computes a 16(M) x 64(N) tile: A fragment reused across 4 WMMAs.
// All B fragments for a K-step are loaded before the first WMMA so loads overlap.
// M%16==0, K%32==0, Nc%64==0 guaranteed by caller. bias always valid.
template <bool HASR, bool RELU, bool WBF>
__global__ void wmma_gemm(const ushort_t* __restrict__ Abf, const ushort_t* __restrict__ Bpk,
                          const float* __restrict__ bias, const float* __restrict__ R,
                          float* __restrict__ C, ushort_t* __restrict__ Cbf,
                          int M, int K, int Nc) {
  const int lane = threadIdx.x & 31;
  const int wave = threadIdx.x >> 5;
  const int colGroups = Nc >> 6;                       // 64-col groups
  const int tiles = (M >> 4) * colGroups;
  const int gt = blockIdx.x * (blockDim.x >> 5) + wave;
  if (gt >= tiles) return;                             // wave-uniform exit (EXEC all-ones)
  const int tm = gt / colGroups, tg = gt % colGroups;
  const int ntiles = Nc >> 4;
  const int nt0 = tg * 4;

  // A fragment per ISA 16-bit 16x32 layout:
  //  lanes 0-15: M=lane,    K = {c..c+7, c+16..c+23} with c=0
  //  lanes16-31: M=lane-16, same with c=8
  const int arow = tm * 16 + (lane & 15);
  const int ac0  = (lane >> 4) * 8;
  const ushort_t* aptr = Abf + (size_t)arow * K + ac0;
  const ushort_t* bp   = Bpk + ((size_t)nt0 * 32 + lane) * 16;
  const size_t bstep   = (size_t)ntiles * 512;         // elements per k-tile

  v8f acc0 = (v8f){0.f,0.f,0.f,0.f,0.f,0.f,0.f,0.f};
  v8f acc1 = acc0, acc2 = acc0, acc3 = acc0;

  for (int kk = 0; kk < K; kk += 32) {
    frag16 a, b0, b1, b2, b3;
    a.q[0]  = *reinterpret_cast<const uint4*>(aptr + kk);        // K=c..c+7
    a.q[1]  = *reinterpret_cast<const uint4*>(aptr + kk + 16);   // K=c+16..c+23
    b0.q[0] = *reinterpret_cast<const uint4*>(bp);
    b0.q[1] = *reinterpret_cast<const uint4*>(bp + 8);
    b1.q[0] = *reinterpret_cast<const uint4*>(bp + 512);
    b1.q[1] = *reinterpret_cast<const uint4*>(bp + 520);
    b2.q[0] = *reinterpret_cast<const uint4*>(bp + 1024);
    b2.q[1] = *reinterpret_cast<const uint4*>(bp + 1032);
    b3.q[0] = *reinterpret_cast<const uint4*>(bp + 1536);
    b3.q[1] = *reinterpret_cast<const uint4*>(bp + 1544);
    __builtin_prefetch(aptr + kk + 32, 0, 1);          // speculative; dropped if OOB
    acc0 = __builtin_amdgcn_wmma_f32_16x16x32_bf16(false, a.v, false, b0.v,
                                                   (short)0, acc0, false, false);
    acc1 = __builtin_amdgcn_wmma_f32_16x16x32_bf16(false, a.v, false, b1.v,
                                                   (short)0, acc1, false, false);
    acc2 = __builtin_amdgcn_wmma_f32_16x16x32_bf16(false, a.v, false, b2.v,
                                                   (short)0, acc2, false, false);
    acc3 = __builtin_amdgcn_wmma_f32_16x16x32_bf16(false, a.v, false, b3.v,
                                                   (short)0, acc3, false, false);
    bp += bstep;
  }

  // C layout: VGPR r -> M = r + 8*(lane>=16), N = lane&15
  const int crow0 = tm * 16 + ((lane >> 4) << 3);
  v8f accs[4] = {acc0, acc1, acc2, acc3};
#pragma unroll
  for (int n = 0; n < 4; ++n) {
    const int ccol = (nt0 + n) * 16 + (lane & 15);
    const float bvv = bias[ccol];
#pragma unroll
    for (int r = 0; r < 8; ++r) {
      size_t idx = (size_t)(crow0 + r) * Nc + ccol;
      float v = accs[n][r] + bvv;
      if (HASR) v += R[idx];
      if (RELU) v = fmaxf(v, 0.f);
      C[idx] = v;
      if (WBF) Cbf[idx] = f2bf(v);
    }
  }
}

// pack weight B[K x Nc] (fp32 row-major) into per-lane WMMA fragments (bf16)
__global__ void pack_b(const float* __restrict__ B, ushort_t* __restrict__ Bpk,
                       int K, int Nc) {
  int t = blockIdx.x * blockDim.x + threadIdx.x;
  int ntiles = Nc >> 4;
  int total = (K >> 5) * ntiles * 32;
  if (t >= total) return;
  int lane = t & 31, tile = t >> 5;
  int kt = tile / ntiles, nt = tile % ntiles;
  int col = nt * 16 + (lane & 15);
  int kb = kt * 32 + ((lane >> 4) * 8);
  ushort_t* o = Bpk + (size_t)t * 16;
#pragma unroll
  for (int j = 0; j < 8; ++j) o[j]     = f2bf(B[(size_t)(kb + j) * Nc + col]);
#pragma unroll
  for (int j = 0; j < 8; ++j) o[8 + j] = f2bf(B[(size_t)(kb + 16 + j) * Nc + col]);
}

// ---------------- elementwise / init ----------------
__global__ void fill_f32(float* p, float v, size_t n) {
  size_t i = (size_t)blockIdx.x * blockDim.x + threadIdx.x;
  if (i < n) p[i] = v;
}
__global__ void fill_i32(int* p, int v, size_t n) {
  size_t i = (size_t)blockIdx.x * blockDim.x + threadIdx.x;
  if (i < n) p[i] = v;
}
__global__ void cvt_bf16(const float* __restrict__ X, ushort_t* __restrict__ Y, size_t n) {
  size_t i = (size_t)blockIdx.x * blockDim.x + threadIdx.x;
  if (i < n) Y[i] = f2bf(X[i]);
}
__global__ void add2_bf(const float* __restrict__ a, const float* __restrict__ b,
                        float* __restrict__ c, ushort_t* __restrict__ cbf, size_t n) {
  size_t i = (size_t)blockIdx.x * blockDim.x + threadIdx.x;
  if (i < n) { float v = a[i] + b[i]; c[i] = v; cbf[i] = f2bf(v); }
}

// ---------------- degrees ----------------
__global__ void deg_accum(const int* __restrict__ ei, float* __restrict__ deg, int E) {
  int e = blockIdx.x * blockDim.x + threadIdx.x;
  if (e < E) atomicAdd(&deg[ei[E + e]], 1.0f);   // col = edge_index[1]
}
__global__ void deg_to_dinv(float* deg, int n) {
  int i = blockIdx.x * blockDim.x + threadIdx.x;
  if (i < n) { float d = deg[i]; deg[i] = d > 0.f ? rsqrtf(d) : 0.f; }
}

// ---------------- GCN ----------------
__global__ void gcn_self(const float* __restrict__ xw, const float* __restrict__ x,
                         const float* __restrict__ dinv, const float* __restrict__ bg,
                         float* __restrict__ hkv, int n) {
  size_t i = (size_t)blockIdx.x * blockDim.x + threadIdx.x;
  if (i >= (size_t)n * DD) return;
  int node = (int)(i >> 7), c = (int)(i & 127);
  float di = dinv[node];
  hkv[i] = xw[i] * di * di + bg[c] + x[i];       // self-loop + bias + residual
}
__global__ void gcn_scatter(const int* __restrict__ ei, const float* __restrict__ xw,
                            const float* __restrict__ dinv, float* __restrict__ hkv, int E) {
  size_t t = (size_t)blockIdx.x * blockDim.x + threadIdx.x;
  int e = (int)(t >> 5), lane = (int)(t & 31);
  if (e >= E) return;
  int row = ei[e], col = ei[E + e];
  float w = dinv[row] * dinv[col];
  const float4 v = *reinterpret_cast<const float4*>(xw + (size_t)row * DD + lane * 4);
  float* dst = hkv + (size_t)col * DD + lane * 4;
  atomicAdd(dst + 0, v.x * w); atomicAdd(dst + 1, v.y * w);
  atomicAdd(dst + 2, v.z * w); atomicAdd(dst + 3, v.w * w);
}

// ---------------- BatchNorm ----------------
#define BN_ROWS 64
__global__ void bn_stats(const float* __restrict__ X, float* __restrict__ stats, int n) {
  int c = threadIdx.x;                           // 128 threads = 128 columns
  int r0 = blockIdx.x * BN_ROWS;
  int rend = r0 + BN_ROWS; if (rend > n) rend = n;
  float s = 0.f, q = 0.f;
  for (int r = r0; r < rend; ++r) {
    float v = X[(size_t)r * DD + c];
    s += v; q += v * v;
  }
  atomicAdd(&stats[c], s);
  atomicAdd(&stats[DD + c], q);
}
__global__ void bn_apply(const float* __restrict__ X, float* __restrict__ Y,
                         const float* __restrict__ stats, const float* __restrict__ g,
                         const float* __restrict__ b, int n) {
  size_t i = (size_t)blockIdx.x * blockDim.x + threadIdx.x;
  if (i >= (size_t)n * DD) return;
  int c = (int)(i & 127);
  float invn = 1.0f / (float)n;
  float m  = stats[c] * invn;
  float va = stats[DD + c] * invn - m * m;
  float rs = rsqrtf(va + 1e-5f);
  Y[i] = (X[i] - m) * rs * g[c] + b[c];
}
__global__ void bn_apply_bf(const float* __restrict__ X, float* __restrict__ Y,
                            ushort_t* __restrict__ Ybf, const float* __restrict__ stats,
                            const float* __restrict__ g, const float* __restrict__ b, int n) {
  size_t i = (size_t)blockIdx.x * blockDim.x + threadIdx.x;
  if (i >= (size_t)n * DD) return;
  int c = (int)(i & 127);
  float invn = 1.0f / (float)n;
  float m  = stats[c] * invn;
  float va = stats[DD + c] * invn - m * m;
  float rs = rsqrtf(va + 1e-5f);
  float v = (X[i] - m) * rs * g[c] + b[c];
  Y[i] = v; Ybf[i] = f2bf(v);
}

// ---------------- attention ----------------
__global__ void attn_scores(const int* __restrict__ ei, const float* __restrict__ Km,
                            const float* __restrict__ x, float* __restrict__ sc,
                            int* __restrict__ smax, int E) {
  int t = blockIdx.x * blockDim.x + threadIdx.x;
  if (t >= E * HH) return;
  int e = t >> 3, h = t & 7;
  int src = ei[e], dst = ei[E + e];
  const float4* kp = reinterpret_cast<const float4*>(Km + (size_t)src * DD + h * HDm);
  const float4* qp = reinterpret_cast<const float4*>(x  + (size_t)dst * DD + h * HDm);
  float s = 0.f;
#pragma unroll
  for (int j = 0; j < 4; ++j) {
    float4 a = kp[j], b = qp[j];
    s += a.x * b.x + a.y * b.y + a.z * b.z + a.w * b.w;
  }
  s *= 0.25f;                                    // 1/sqrt(16)
  sc[t] = s;
  atomicMax(&smax[dst * HH + h], f2ord(s));
}
__global__ void attn_weights(const int* __restrict__ ei, float* __restrict__ sc,
                             const int* __restrict__ smax, float* __restrict__ denom, int E) {
  int t = blockIdx.x * blockDim.x + threadIdx.x;
  if (t >= E * HH) return;
  int e = t >> 3, h = t & 7;
  int dst = ei[E + e];
  float w = __expf(sc[t] - ord2f(smax[dst * HH + h]));
  sc[t] = w;
  atomicAdd(&denom[dst * HH + h], w);
}
__global__ void attn_scatter(const int* __restrict__ ei, const float* __restrict__ sc,
                             const float* __restrict__ Vm, float* __restrict__ att, int E) {
  size_t t = (size_t)blockIdx.x * blockDim.x + threadIdx.x;
  int e = (int)(t >> 5), lane = (int)(t & 31);
  if (e >= E) return;
  int src = ei[e], dst = ei[E + e];
  int h = lane >> 2, j0 = (lane & 3) * 4;
  float w = sc[(size_t)e * HH + h];
  const float4 v = *reinterpret_cast<const float4*>(Vm + (size_t)src * DD + h * HDm + j0);
  float* dp = att + (size_t)dst * DD + h * HDm + j0;
  atomicAdd(dp + 0, v.x * w); atomicAdd(dp + 1, v.y * w);
  atomicAdd(dp + 2, v.z * w); atomicAdd(dp + 3, v.w * w);
}
__global__ void attn_div_bf(const float* __restrict__ att, const float* __restrict__ denom,
                            ushort_t* __restrict__ abf, int n) {
  size_t i = (size_t)blockIdx.x * blockDim.x + threadIdx.x;
  if (i >= (size_t)n * DD) return;
  int node = (int)(i >> 7), h = (int)((i & 127) >> 4);
  abf[i] = f2bf(att[i] / (denom[node * HH + h] + 1e-16f));
}

// ---------------- driver ----------------
extern "C" void kernel_launch(void* const* d_in, const int* in_sizes, int n_in,
                              void* d_out, int out_size, void* d_ws, size_t ws_size,
                              hipStream_t stream) {
  const float* x     = (const float*)d_in[0];
  const int*   ei    = (const int*)  d_in[1];
  const float* W_gcn = (const float*)d_in[2];
  const float* b_gcn = (const float*)d_in[3];
  const float* Wk    = (const float*)d_in[4];
  const float* bk    = (const float*)d_in[5];
  const float* Wv    = (const float*)d_in[6];
  const float* bv    = (const float*)d_in[7];
  const float* Wo    = (const float*)d_in[8];
  const float* bo    = (const float*)d_in[9];
  const float* g1    = (const float*)d_in[10];
  const float* be1   = (const float*)d_in[11];
  const float* g2    = (const float*)d_in[12];
  const float* be2   = (const float*)d_in[13];
  const float* g3    = (const float*)d_in[14];
  const float* be3   = (const float*)d_in[15];
  const float* W1    = (const float*)d_in[16];
  const float* b1    = (const float*)d_in[17];
  const float* W2    = (const float*)d_in[18];
  const float* b2    = (const float*)d_in[19];

  const int N = in_sizes[0] / DD;
  const int E = in_sizes[1] / 2;
  const size_t ND = (size_t)N * DD;

  // ---- workspace layout ----
  float* ws    = (float*)d_ws;
  float* deg   = ws;                       // N
  float* hkv   = deg + N;                  // N*D
  float* xw    = hkv + ND;                 // N*D (x@Wgcn; later h_attn)
  float* Km    = xw + ND;                  // N*D (K; later FFN out)
  float* Vm    = Km + ND;                  // N*D
  float* att   = Vm + ND;                  // N*D (attn accum; later h_comb)
  float* mid   = att + ND;                 // N*2D
  float* scw   = mid + (size_t)N * DD2;    // E*H
  int*   smax  = (int*)(scw + (size_t)E * HH);     // N*H
  float* denom = (float*)(smax + (size_t)N * HH);  // N*H
  float* stats = denom + (size_t)N * HH;   // 2*D
  float* zerob = stats + 2 * DD;           // 256 (zero bias)
  // bf16 region
  ushort_t* xbf   = (ushort_t*)(zerob + DD2);     // N*D
  ushort_t* hkvbf = xbf + ND;                     // N*D
  ushort_t* abf1  = hkvbf + ND;                   // N*D  (att bf -> hcomb bf)
  ushort_t* midbf = abf1 + ND;                    // N*2D
  ushort_t* pkg   = midbf + (size_t)N * DD2;      // packed weights:
  ushort_t* pWg   = pkg;                          //  128*128
  ushort_t* pWk   = pWg + DD * DD;
  ushort_t* pWv   = pWk + DD * DD;
  ushort_t* pWo   = pWv + DD * DD;
  ushort_t* pW1   = pWo + DD * DD;                //  128*256
  ushort_t* pW2   = pW1 + DD * DD2;               //  256*128

  auto blks = [](size_t n) { return (unsigned)((n + 255) / 256); };
  auto pack = [&](const float* B, ushort_t* P, int K, int Nc) {
    size_t tot = (size_t)(K >> 5) * (Nc >> 4) * 32;
    pack_b<<<blks(tot), 256, 0, stream>>>(B, P, K, Nc);
  };
  auto gemmGrid = [&](int M, int Nc) { return (unsigned)(((M / 16) * (Nc / 64) + 7) / 8); };

  // ---- prep: bf16 conversions + weight packing + zero bias ----
  cvt_bf16<<<blks(ND), 256, 0, stream>>>(x, xbf, ND);
  pack(W_gcn, pWg, DD, DD);  pack(Wk, pWk, DD, DD);  pack(Wv, pWv, DD, DD);
  pack(Wo, pWo, DD, DD);     pack(W1, pW1, DD, DD2); pack(W2, pW2, DD2, DD);
  fill_f32<<<1, DD2, 0, stream>>>(zerob, 0.f, (size_t)DD2);

  // ---- GCN expert ----
  fill_f32<<<blks(N), 256, 0, stream>>>(deg, 1.0f, (size_t)N);      // self-loop
  deg_accum<<<blks(E), 256, 0, stream>>>(ei, deg, E);
  deg_to_dinv<<<blks(N), 256, 0, stream>>>(deg, N);
  wmma_gemm<false, false, false><<<gemmGrid(N, DD), 256, 0, stream>>>(
      xbf, pWg, zerob, nullptr, xw, nullptr, N, DD, DD);
  gcn_self<<<blks(ND), 256, 0, stream>>>(xw, x, deg, b_gcn, hkv, N);
  gcn_scatter<<<blks((size_t)E * 32), 256, 0, stream>>>(ei, xw, deg, hkv, E);
  // BN1 (also emit bf16 copy of h_kv)
  fill_f32<<<1, 256, 0, stream>>>(stats, 0.f, (size_t)2 * DD);
  bn_stats<<<(N + BN_ROWS - 1) / BN_ROWS, DD, 0, stream>>>(hkv, stats, N);
  bn_apply_bf<<<blks(ND), 256, 0, stream>>>(hkv, hkv, hkvbf, stats, g1, be1, N);

  // ---- attention ----
  wmma_gemm<false, false, false><<<gemmGrid(N, DD), 256, 0, stream>>>(
      hkvbf, pWk, bk, nullptr, Km, nullptr, N, DD, DD);
  wmma_gemm<false, false, false><<<gemmGrid(N, DD), 256, 0, stream>>>(
      hkvbf, pWv, bv, nullptr, Vm, nullptr, N, DD, DD);
  fill_i32<<<blks((size_t)N * HH), 256, 0, stream>>>(smax, (int)0x807FFFFF, (size_t)N * HH);
  fill_f32<<<blks((size_t)N * HH), 256, 0, stream>>>(denom, 0.f, (size_t)N * HH);
  fill_f32<<<blks(ND), 256, 0, stream>>>(att, 0.f, ND);
  attn_scores <<<blks((size_t)E * HH), 256, 0, stream>>>(ei, Km, x, scw, smax, E);
  attn_weights<<<blks((size_t)E * HH), 256, 0, stream>>>(ei, scw, smax, denom, E);
  attn_scatter<<<blks((size_t)E * 32), 256, 0, stream>>>(ei, scw, Vm, att, E);
  attn_div_bf <<<blks(ND), 256, 0, stream>>>(att, denom, abf1, N);
  wmma_gemm<true, false, false><<<gemmGrid(N, DD), 256, 0, stream>>>(
      abf1, pWo, bo, x, xw, nullptr, N, DD, DD);                    // +x residual
  // BN2
  fill_f32<<<1, 256, 0, stream>>>(stats, 0.f, (size_t)2 * DD);
  bn_stats<<<(N + BN_ROWS - 1) / BN_ROWS, DD, 0, stream>>>(xw, stats, N);
  bn_apply<<<blks(ND), 256, 0, stream>>>(xw, xw, stats, g2, be2, N);

  // ---- combine + FFN ----
  add2_bf<<<blks(ND), 256, 0, stream>>>(hkv, xw, att, abf1, ND);    // h = h_kv + h_attn
  wmma_gemm<false, true, true><<<gemmGrid(N, DD2), 256, 0, stream>>>(
      abf1, pW1, b1, nullptr, mid, midbf, N, DD, DD2);              // relu(h@W1+b1)
  wmma_gemm<true, false, false><<<gemmGrid(N, DD), 256, 0, stream>>>(
      midbf, pW2, b2, att, Km, nullptr, N, DD2, DD);                // h + mid@W2+b2
  // BN3 -> output
  fill_f32<<<1, 256, 0, stream>>>(stats, 0.f, (size_t)2 * DD);
  bn_stats<<<(N + BN_ROWS - 1) / BN_ROWS, DD, 0, stream>>>(Km, stats, N);
  bn_apply<<<blks(ND), 256, 0, stream>>>(Km, (float*)d_out, stats, g3, be3, N);
}
